// XLNet_83691732730426
// MI455X (gfx1250) — matrix operand order repaired and checked by reading, not compiled, not run on hardware
//
#include <hip/hip_runtime.h>
#include <cstdint>
#include <cstddef>

// ------------------------------------------------------------------
// Fixed problem dimensions (XLNet two-stream rel-attention layer)
// ------------------------------------------------------------------
#define QL   512    // QLEN
#define MLN  512    // MLEN
#define BS   2      // BSZ
#define HIDN 512    // HID
#define NHD  8      // NHEAD
#define DH   512    // DHEAD
#define NPD  128    // NPRED
#define KLN  1024   // KLEN = QLEN + MLEN
#define RL   1025   // RLEN = KLEN + 1
#define NDX  4096   // NHEAD * DHEAD
#define SCL  0.044194173824159216f  // 1/sqrt(512)

typedef unsigned short u16;
typedef __attribute__((ext_vector_type(16))) __bf16 v16bf;
typedef __attribute__((ext_vector_type(8)))  float  v8f;

#ifndef __has_builtin
#define __has_builtin(x) 0
#endif

#if __has_builtin(__builtin_amdgcn_global_load_async_to_lds_b128)
#define ASYNC_LDS 1
#else
#define ASYNC_LDS 0
#endif

#if ASYNC_LDS
// Builtin signature (from hipcc diagnostic): param0 = int4-vector pointer in
// the global address space (AS1, printed as `__device__` in HIP), param1 =
// int4-vector pointer in LDS (AS3), then imm offset / imm cpol.
typedef int v4i __attribute__((ext_vector_type(4)));
typedef __attribute__((address_space(1))) v4i gv4i;
typedef __attribute__((address_space(3))) v4i lv4i;
__device__ __forceinline__ void async_copy32(const u16* gsrc, u16* ldst) {
  gv4i* g = (gv4i*)gsrc;
  lv4i* l = (lv4i*)ldst;
  // two B128 async copies (same imm offset applies to both global & LDS sides)
  __builtin_amdgcn_global_load_async_to_lds_b128(g, l, 0, 0);
  __builtin_amdgcn_global_load_async_to_lds_b128(g, l, 16, 0);
}
__device__ __forceinline__ void async_wait0() {
#if __has_builtin(__builtin_amdgcn_s_wait_asynccnt)
  __builtin_amdgcn_s_wait_asynccnt(0);
#else
  asm volatile("s_wait_asynccnt 0" ::: "memory");
#endif
}
#endif

__device__ __forceinline__ u16 f2bf(float f) {
  unsigned u = __float_as_uint(f);
  u += 0x7FFFu + ((u >> 16) & 1u);     // round-to-nearest-even
  return (u16)(u >> 16);
}
__device__ __forceinline__ float bf2f(u16 h) {
  return __uint_as_float(((unsigned)h) << 16);
}

union Frag { v16bf v; uint4 q[2]; unsigned u[8]; };

// ------------------------------------------------------------------
// Batched bf16 WMMA GEMM (double-buffered, async global->LDS staging).
//   BT=true : C[M,N] = A[M,K] * B[N,K]^T   (both row-major)
//   BT=false: C[M,N] = A[M,K] * B[K,N]
//   OBF     : output bf16 (else f32). Optional bias[N] added.
// Tile 128x128, 256 threads (8 wave32 waves, 4x2), K-step 32.
// LDS tiles [128][40] u16: 80B row pitch -> rows 16B aligned (async B128 ok)
// and bank phase 20*m mod 64 tiles all 64 banks -> conflict-free ds_load_b128.
// Fragment layouts per CDNA5 ISA 7.12.2:
//   A 16x32 bf16: lanes 0-15 -> M=l, K {0..7,16..23}; lanes 16-31 -> {8..15,24..31}
//   B 32x16 bf16: lanes 0-15 -> N=l, K {0..7,16..23}; lanes 16-31 -> {8..15,24..31}
// M/N edges handled by clamping load rows (tail garbage never stored).
// ------------------------------------------------------------------
#define LPITCH 40

template<bool BT, bool OBF>
__global__ __launch_bounds__(256)
void gemm_wmma(const u16* __restrict__ A, int lda, long long sA,
               const u16* __restrict__ Bm, int ldb, long long sB,
               void* __restrict__ Cv, int ldc, long long sC,
               const float* __restrict__ bias, int M, int N, int K)
{
  __shared__ alignas(16) u16 As[2][128 * LPITCH];
  __shared__ alignas(16) u16 Bs[2][128 * LPITCH];

  const int z = blockIdx.z;
  A  += (long long)z * sA;
  Bm += (long long)z * sB;
  const long long cOff = (long long)z * sC;

  const int m0 = blockIdx.y * 128;
  const int n0 = blockIdx.x * 128;
  const int t    = threadIdx.x;
  const int lane = t & 31;
  const int wave = t >> 5;
  const int wm = wave >> 1;   // 0..3 : 32-row strip
  const int wn = wave & 1;    // 0..1 : 64-col strip

  v8f acc[2][4];
  #pragma unroll
  for (int i = 0; i < 2; ++i)
    #pragma unroll
    for (int j = 0; j < 4; ++j)
      #pragma unroll
      for (int r = 0; r < 8; ++r) acc[i][j][r] = 0.0f;

  const int arow = t >> 1;         // 0..127
  const int acol = (t & 1) * 16;   // 0 | 16
  const int bk   = t >> 3;         // 0..31   (NN loader)
  const int bn   = (t & 7) * 16;   // 0..112  (NN loader)

  int agr = m0 + arow; if (agr >= M) agr = M - 1;   // clamped A row
  const u16* aRow = A + (long long)agr * lda + acol;

  int bgr = n0 + arow; if (bgr >= N) bgr = N - 1;   // clamped B^T row
  const u16* bRow = BT ? (Bm + (long long)bgr * ldb + acol) : nullptr;

  auto load_stage = [&](int buf, int k0) {
    { // ---- A tile: rows m0..m0+127, cols k0..k0+31 ----
      const u16* gsrc = aRow + k0;
      u16* ldst = &As[buf][arow * LPITCH + acol];
#if ASYNC_LDS
      async_copy32(gsrc, ldst);
#else
      const uint4* s4 = (const uint4*)gsrc;
      uint4 p0 = s4[0], p1 = s4[1];
      *(uint4*)ldst = p0; *(uint4*)(ldst + 8) = p1;
#endif
    }
    if (BT) { // ---- B tile (B row-major [N,K]): direct rows ----
      const u16* gsrc = bRow + k0;
      u16* ldst = &Bs[buf][arow * LPITCH + acol];
#if ASYNC_LDS
      async_copy32(gsrc, ldst);
#else
      const uint4* s4 = (const uint4*)gsrc;
      uint4 p0 = s4[0], p1 = s4[1];
      *(uint4*)ldst = p0; *(uint4*)(ldst + 8) = p1;
#endif
    } else {  // ---- B tile (B row-major [K,N]): transpose into Bs[n][k] ----
      const int gk = k0 + bk;
      const int gn = n0 + bn;
      union { uint4 q[2]; u16 s[16]; } v;
      if (gn + 16 <= N) {
        const uint4* src = (const uint4*)(Bm + (long long)gk * ldb + gn);
        v.q[0] = src[0]; v.q[1] = src[1];
      } else {
        #pragma unroll
        for (int j = 0; j < 16; ++j)
          v.s[j] = (gn + j < N) ? Bm[(long long)gk * ldb + gn + j] : (u16)0;
      }
      #pragma unroll
      for (int j = 0; j < 16; ++j) Bs[buf][(bn + j) * LPITCH + bk] = v.s[j];
    }
  };

  const int nk = K >> 5;            // K is a multiple of 32 for every call site
  load_stage(0, 0);

  const int lh  = lane & 15;
  const int ka  = (lane < 16) ? 0 : 8;    // A-frag K base (u16 units)
  const int kbb = (lane < 16) ? 0 : 8;    // B-frag K base (u16 units)

  for (int ks = 0; ks < nk; ++ks) {
    const int cur = ks & 1;
#if ASYNC_LDS
    async_wait0();                  // my slice of stage `ks` landed in LDS
#endif
    __syncthreads();                // everyone's slice visible
    if (ks + 1 < nk) load_stage(cur ^ 1, (ks + 1) << 5);  // overlap with compute

    Frag a[2], b[4];
    #pragma unroll
    for (int mi = 0; mi < 2; ++mi) {
      const u16* p = &As[cur][(wm * 32 + mi * 16 + lh) * LPITCH];
      a[mi].q[0] = *(const uint4*)(p + ka);        // 16B, conflict-free
      a[mi].q[1] = *(const uint4*)(p + ka + 16);
    }
    #pragma unroll
    for (int ni = 0; ni < 4; ++ni) {
      const u16* p = &Bs[cur][(wn * 64 + ni * 16 + lh) * LPITCH];
      b[ni].q[0] = *(const uint4*)(p + kbb);
      b[ni].q[1] = *(const uint4*)(p + kbb + 16);
    }
    #pragma unroll
    for (int mi = 0; mi < 2; ++mi)
      #pragma unroll
      for (int ni = 0; ni < 4; ++ni)
        acc[mi][ni] = __builtin_amdgcn_wmma_f32_16x16x32_bf16(
            false, a[mi].v, false, b[ni].v, (short)0, acc[mi][ni], false, false);
  }

  // ---- store: C layout per ISA (VGPR r: lanes 0-15 M=r, lanes 16-31 M=8+r) ----
  float* Cf = (float*)Cv;
  u16*   Cb = (u16*)Cv;
  const int rb = (lane >> 4) * 8;
  #pragma unroll
  for (int mi = 0; mi < 2; ++mi) {
    const int rbase = m0 + wm * 32 + mi * 16 + rb;
    #pragma unroll
    for (int ni = 0; ni < 4; ++ni) {
      const int cn = n0 + wn * 64 + ni * 16 + lh;
      if (cn < N) {
        const float ba = bias ? bias[cn] : 0.0f;
        #pragma unroll
        for (int r = 0; r < 8; ++r) {
          const int row = rbase + r;
          if (row < M) {
            const float vv = acc[mi][ni][r] + ba;
            const long long idx = cOff + (long long)row * ldc + cn;
            if (OBF) Cb[idx] = f2bf(vv); else Cf[idx] = vv;
          }
        }
      }
    }
  }
}

// ------------------------------------------------------------------
// Elementwise / prep kernels
// ------------------------------------------------------------------
__global__ void k_cvt(const float* __restrict__ in, u16* __restrict__ out, long long n) {
  long long i = (long long)blockIdx.x * 256 + threadIdx.x;
  if (i < n) out[i] = f2bf(in[i]);
}

__global__ void k_concat(const float* __restrict__ mems, const float* __restrict__ h,
                         u16* __restrict__ out) {
  long long i = (long long)blockIdx.x * 256 + threadIdx.x;
  const long long half = (long long)MLN * BS * HIDN;
  const long long tot  = half + (long long)QL * BS * HIDN;
  if (i < tot) out[i] = f2bf(i < half ? mems[i] : h[i - half]);
}

// [HIDN, NDX] f32 -> [NDX, HIDN] bf16
__global__ void k_wtrans(const float* __restrict__ in, u16* __restrict__ out) {
  long long i = (long long)blockIdx.x * 256 + threadIdx.x;
  const long long tot = (long long)HIDN * NDX;
  if (i < tot) {
    int r = (int)(i / NDX), c = (int)(i % NDX);
    out[(long long)c * HIDN + r] = f2bf(in[i]);
  }
}

// target_mapping [QL,NPD,BS] -> tm_bf [BS,QL,NPD], tmT_bf [BS,NPD,QL]
__global__ void k_prep_tm(const float* __restrict__ tm,
                          u16* __restrict__ tm_bf, u16* __restrict__ tmT) {
  long long i = (long long)blockIdx.x * 256 + threadIdx.x;
  const long long tot = (long long)QL * NPD * BS;
  if (i >= tot) return;
  int b = (int)(i % BS);
  int l = (int)((i / BS) % NPD);
  int m = (int)(i / ((long long)BS * NPD));
  u16 v = f2bf(tm[i]);
  tm_bf[((long long)b * QL + m) * NPD + l] = v;
  tmT  [((long long)b * NPD + l) * QL + m] = v;
}

// seg_mat [QL,KLN,BS,2] f32 -> [BS,QL,2*KLN] bf16 (j*2+x minor)
__global__ void k_prep_seg(const float* __restrict__ seg, u16* __restrict__ out) {
  long long i = (long long)blockIdx.x * 256 + threadIdx.x;
  const long long tot = (long long)QL * KLN * BS * 2;
  if (i >= tot) return;
  int x = (int)(i & 1);
  int b = (int)((i >> 1) % BS);
  int j = (int)((i >> 1) / BS % KLN);
  int m = (int)(i / (2LL * BS * KLN));
  out[((long long)b * QL + m) * (2 * KLN) + j * 2 + x] = f2bf(seg[i]);
}

// [rows,NDX] bf16 (+ optional f32 bias[NDX]) -> [BS,NHD,S,DH] bf16
__global__ void k_perm_to_heads(const u16* __restrict__ in, const float* __restrict__ bias,
                                u16* __restrict__ out, int S, int seq_major) {
  long long i = (long long)blockIdx.x * 256 + threadIdx.x;
  const long long tot = (long long)BS * NHD * S * DH;
  if (i >= tot) return;
  int d = (int)(i % DH);
  int s = (int)((i / DH) % S);
  int n = (int)((i / ((long long)DH * S)) % NHD);
  int b = (int)(i / ((long long)DH * S * NHD));
  long long row = seq_major ? ((long long)s * BS + b) : ((long long)b * S + s);
  float v = bf2f(in[row * NDX + n * DH + d]);
  if (bias) v += bias[n * DH + d];
  out[i] = f2bf(v);
}

// [BS,NHD,S,DH] bf16 -> [BS,S,NDX] bf16
__global__ void k_perm_from_heads(const u16* __restrict__ in, u16* __restrict__ out, int S) {
  long long i = (long long)blockIdx.x * 256 + threadIdx.x;
  const long long tot = (long long)BS * NHD * S * DH;
  if (i >= tot) return;
  int d = (int)(i % DH);
  int n = (int)((i / DH) % NHD);
  int s = (int)((i / ((long long)DH * NHD)) % S);
  int b = (int)(i / ((long long)DH * NHD * S));
  out[i] = in[(((long long)b * NHD + n) * S + s) * DH + d];
}

// ef1[b,n,s,x] = sum_d (q[s,b,n,d] + r_s_bias[n,d]) * seg_embed[x,n,d]
__global__ void k_ef1(const u16* __restrict__ q, const float* __restrict__ rs,
                      const float* __restrict__ se, float* __restrict__ ef1,
                      int S, int seq_major) {
  int i = blockIdx.x * 256 + threadIdx.x;
  const int tot = BS * NHD * S * 2;
  if (i >= tot) return;
  int x = i & 1;
  int s = (i >> 1) % S;
  int n = (i >> 1) / S % NHD;
  int b = (i >> 1) / (S * NHD);
  long long row = seq_major ? ((long long)s * BS + b) : ((long long)b * S + s);
  const u16*   qp = q + row * NDX + n * DH;
  const float* rp = rs + n * DH;
  const float* sp = se + ((long long)x * NHD + n) * DH;
  float acc = 0.0f;
  for (int d = 0; d < DH; ++d) acc += (bf2f(qp[d]) + rp[d]) * sp[d];
  ef1[i] = acc;
}

// masked softmax row: score = (ac + rel_shift(bd) + ef)*SCL - 1e30*mask
__global__ __launch_bounds__(256)
void k_softmax(const float* __restrict__ ac, const float* __restrict__ bd,
               const float* __restrict__ ef1, const u16* __restrict__ seg,
               const float* __restrict__ mask, u16* __restrict__ prob, int S) {
  const int iq = blockIdx.x, n = blockIdx.y, b = blockIdx.z, t = threadIdx.x;
  const long long bn = (long long)b * NHD + n;
  const float* acr = ac + (bn * S + iq) * (long long)KLN;
  const float* bdr = bd + bn * (long long)S * RL;
  const float  e0  = ef1[(bn * S + iq) * 2];
  const float  e1  = ef1[(bn * S + iq) * 2 + 1];
  const u16*   sgr = seg + ((long long)b * S + iq) * (2 * KLN);
  const float* mr  = mask + (long long)iq * KLN * BS + b;

  float sc[4];
  float mx = -3.4e38f;
  #pragma unroll
  for (int u = 0; u < 4; ++u) {
    const int j = t + u * 256;
    const long long f = (long long)S + (long long)iq * (RL - 1) + j; // rel_shift remap
    const int i2 = (int)(f / RL), j2 = (int)(f % RL);
    const float bdv = bdr[(long long)i2 * RL + j2];
    const float efv = bf2f(sgr[2 * j]) * e0 + bf2f(sgr[2 * j + 1]) * e1;
    const float s = (acr[j] + bdv + efv) * SCL - 1e30f * mr[(long long)j * BS];
    sc[u] = s;
    mx = fmaxf(mx, s);
  }
  __shared__ float red[256];
  red[t] = mx; __syncthreads();
  for (int st = 128; st > 0; st >>= 1) { if (t < st) red[t] = fmaxf(red[t], red[t + st]); __syncthreads(); }
  mx = red[0]; __syncthreads();

  float sm = 0.0f;
  #pragma unroll
  for (int u = 0; u < 4; ++u) { sc[u] = __expf(sc[u] - mx); sm += sc[u]; }
  red[t] = sm; __syncthreads();
  for (int st = 128; st > 0; st >>= 1) { if (t < st) red[t] += red[t + st]; __syncthreads(); }
  sm = red[0];

  const float inv = 1.0f / sm;
  u16* pr = prob + (bn * S + iq) * (long long)KLN;
  #pragma unroll
  for (int u = 0; u < 4; ++u) pr[t + u * 256] = f2bf(sc[u] * inv);
}

// LayerNorm(attn[b,i,:] + res[i,b,:]) -> out[i,b,:]
__global__ __launch_bounds__(256)
void k_ln(const float* __restrict__ attn, const float* __restrict__ res,
          const float* __restrict__ gamma, const float* __restrict__ beta,
          float* __restrict__ out, int S) {
  const int rid = blockIdx.x, t = threadIdx.x;
  const int i = rid / BS, b = rid % BS;
  const float* ar = attn + ((long long)b * S + i) * HIDN;
  const float* rr = res  + ((long long)i * BS + b) * HIDN;
  const float x0 = ar[t] + rr[t];
  const float x1 = ar[t + 256] + rr[t + 256];
  __shared__ float red[256];
  red[t] = x0 + x1; __syncthreads();
  for (int st = 128; st > 0; st >>= 1) { if (t < st) red[t] += red[t + st]; __syncthreads(); }
  const float mu = red[0] * (1.0f / HIDN); __syncthreads();
  const float d0 = x0 - mu, d1 = x1 - mu;
  red[t] = d0 * d0 + d1 * d1; __syncthreads();
  for (int st = 128; st > 0; st >>= 1) { if (t < st) red[t] += red[t + st]; __syncthreads(); }
  const float rstd = rsqrtf(red[0] * (1.0f / HIDN) + 1e-5f);
  float* o = out + ((long long)i * BS + b) * HIDN;
  o[t]       = d0 * rstd * gamma[t] + beta[t];
  o[t + 256] = d1 * rstd * gamma[t + 256] + beta[t + 256];
}

// ------------------------------------------------------------------
// Host orchestration
// ------------------------------------------------------------------
extern "C" void kernel_launch(void* const* d_in, const int* in_sizes, int n_in,
                              void* d_out, int out_size, void* d_ws, size_t ws_size,
                              hipStream_t stream) {
  (void)in_sizes; (void)n_in; (void)out_size; (void)ws_size;

  const float* h_in    = (const float*)d_in[0];
  const float* g_in    = (const float*)d_in[1];
  const float* r_in    = (const float*)d_in[2];
  const float* mems    = (const float*)d_in[3];
  const float* rw_b    = (const float*)d_in[4];
  const float* rr_b    = (const float*)d_in[5];
  const float* rs_b    = (const float*)d_in[6];
  const float* se      = (const float*)d_in[7];
  const float* segm    = (const float*)d_in[8];
  const float* mask_h  = (const float*)d_in[9];
  const float* mask_g  = (const float*)d_in[10];
  const float* tmap    = (const float*)d_in[11];
  const float* Wq      = (const float*)d_in[12];
  const float* Wk      = (const float*)d_in[13];
  const float* Wv      = (const float*)d_in[14];
  const float* Wr      = (const float*)d_in[15];
  const float* Wo      = (const float*)d_in[16];
  const float* key_w   = (const float*)d_in[17]; const float* key_b   = (const float*)d_in[18];
  const float* val_w   = (const float*)d_in[19]; const float* val_b   = (const float*)d_in[20];
  const float* qlin_w  = (const float*)d_in[21]; const float* qlin_b  = (const float*)d_in[22];
  const float* rlin_w  = (const float*)d_in[23]; const float* rlin_b  = (const float*)d_in[24];
  const float* glin_w  = (const float*)d_in[25]; const float* glin_b  = (const float*)d_in[26];
  const float* dense_w = (const float*)d_in[27]; const float* dense_b = (const float*)d_in[28];
  const float* ln_gam  = (const float*)d_in[29]; const float* ln_bet  = (const float*)d_in[30];
  float* out = (float*)d_out;

  // ---- workspace bump allocator ----
  char* wp = (char*)d_ws;
  auto au16 = [&](long long n) -> u16* { u16* p = (u16*)wp; wp += ((n * 2 + 255) / 256) * 256; return p; };
  auto af32 = [&](long long n) -> float* { float* p = (float*)wp; wp += ((n * 4 + 255) / 256) * 256; return p; };

  u16* h_bf   = au16((long long)QL * BS * HIDN);
  u16* g_bf   = au16((long long)QL * BS * HIDN);
  u16* r_bf   = au16((long long)RL * BS * HIDN);
  u16* cat_bf = au16((long long)KLN * BS * HIDN);
  u16* WqT = au16((long long)NDX * HIDN);
  u16* WkT = au16((long long)NDX * HIDN);
  u16* WvT = au16((long long)NDX * HIDN);
  u16* WrT = au16((long long)NDX * HIDN);
  u16* w_key = au16(DH * DH); u16* w_val = au16(DH * DH);
  u16* w_ql  = au16(DH * DH); u16* w_rl  = au16(DH * DH);
  u16* w_gl  = au16(DH * DH); u16* w_dn  = au16(DH * DH);
  u16* w_po  = au16((long long)HIDN * NDX);
  u16* tm_bf  = au16((long long)BS * QL * NPD);
  u16* tmT_bf = au16((long long)BS * NPD * QL);
  u16* seg_h  = au16((long long)BS * QL * 2 * KLN);
  u16* seg_g  = au16((long long)BS * NPD * 2 * KLN);
  u16* Kh = au16((long long)BS * NHD * KLN * DH);
  u16* Vh = au16((long long)BS * NHD * KLN * DH);
  u16* Kr = au16((long long)BS * NHD * RL * DH);
  u16* Qh_rw = au16((long long)BS * NHD * QL * DH);
  u16* Qh_rr = au16((long long)BS * NHD * QL * DH);
  u16* Qg_rw = au16((long long)BS * NHD * NPD * DH);
  u16* Qg_rr = au16((long long)BS * NHD * NPD * DH);
  u16* qgath = au16((long long)BS * NPD * NDX);
  u16* tmp0  = au16((long long)RL * BS * NDX);
  u16* tmp1  = au16((long long)RL * BS * NDX);
  float* ac = af32((long long)BS * NHD * QL * KLN);
  float* bd = af32((long long)BS * NHD * QL * RL);
  u16* prob = au16((long long)BS * NHD * QL * KLN);  // also reused for av3 / av_full
  float* ef1h = af32((long long)BS * NHD * QL * 2);
  float* ef1g = af32((long long)BS * NHD * NPD * 2);
  float* ao = af32((long long)BS * QL * HIDN);

  auto g1 = [](long long n) { return dim3((unsigned)((n + 255) / 256)); };
  auto cvt = [&](const float* s, u16* d, long long n) {
    k_cvt<<<g1(n), 256, 0, stream>>>(s, d, n);
  };
  auto gNTbf = [&](const u16* A, int lda, long long sA, const u16* B, int ldb, long long sB,
                   void* C, int ldc, long long sC, const float* bias, int M, int N, int K, int bt) {
    dim3 g((N + 127) / 128, (M + 127) / 128, bt);
    gemm_wmma<true, true><<<g, 256, 0, stream>>>(A, lda, sA, B, ldb, sB, C, ldc, sC, bias, M, N, K);
  };
  auto gNTf = [&](const u16* A, int lda, long long sA, const u16* B, int ldb, long long sB,
                  void* C, int ldc, long long sC, const float* bias, int M, int N, int K, int bt) {
    dim3 g((N + 127) / 128, (M + 127) / 128, bt);
    gemm_wmma<true, false><<<g, 256, 0, stream>>>(A, lda, sA, B, ldb, sB, C, ldc, sC, bias, M, N, K);
  };
  auto gNNbf = [&](const u16* A, int lda, long long sA, const u16* B, int ldb, long long sB,
                   void* C, int ldc, long long sC, const float* bias, int M, int N, int K, int bt) {
    dim3 g((N + 127) / 128, (M + 127) / 128, bt);
    gemm_wmma<false, true><<<g, 256, 0, stream>>>(A, lda, sA, B, ldb, sB, C, ldc, sC, bias, M, N, K);
  };

  // ---- conversions / prep ----
  cvt(h_in, h_bf, (long long)QL * BS * HIDN);
  cvt(g_in, g_bf, (long long)QL * BS * HIDN);
  cvt(r_in, r_bf, (long long)RL * BS * HIDN);
  k_concat<<<g1((long long)KLN * BS * HIDN), 256, 0, stream>>>(mems, h_in, cat_bf);
  k_wtrans<<<g1((long long)HIDN * NDX), 256, 0, stream>>>(Wq, WqT);
  k_wtrans<<<g1((long long)HIDN * NDX), 256, 0, stream>>>(Wk, WkT);
  k_wtrans<<<g1((long long)HIDN * NDX), 256, 0, stream>>>(Wv, WvT);
  k_wtrans<<<g1((long long)HIDN * NDX), 256, 0, stream>>>(Wr, WrT);
  cvt(key_w, w_key, DH * DH); cvt(val_w, w_val, DH * DH);
  cvt(qlin_w, w_ql, DH * DH); cvt(rlin_w, w_rl, DH * DH);
  cvt(glin_w, w_gl, DH * DH); cvt(dense_w, w_dn, DH * DH);
  cvt(Wo, w_po, (long long)HIDN * NDX);
  k_prep_tm<<<g1((long long)QL * NPD * BS), 256, 0, stream>>>(tmap, tm_bf, tmT_bf);
  k_prep_seg<<<g1((long long)QL * KLN * BS * 2), 256, 0, stream>>>(segm, seg_h);

  const long long headsKL = (long long)BS * NHD * KLN * DH;
  const long long headsQ  = (long long)BS * NHD * QL * DH;
  const long long headsR  = (long long)BS * NHD * RL * DH;
  const long long headsNP = (long long)BS * NHD * NPD * DH;

  // ---- K head: cat x WkT -> key_w linear -> [B,NH,KLN,DH] ----
  gNTbf(cat_bf, HIDN, 0, WkT, HIDN, 0, tmp0, NDX, 0, nullptr, KLN * BS, NDX, HIDN, 1);
  gNTbf(tmp0, DH, 0, w_key, DH, 0, tmp1, DH, 0, key_b, KLN * BS * NHD, DH, DH, 1);
  k_perm_to_heads<<<g1(headsKL), 256, 0, stream>>>(tmp1, nullptr, Kh, KLN, 1);
  // ---- V head ----
  gNTbf(cat_bf, HIDN, 0, WvT, HIDN, 0, tmp0, NDX, 0, nullptr, KLN * BS, NDX, HIDN, 1);
  gNTbf(tmp0, DH, 0, w_val, DH, 0, tmp1, DH, 0, val_b, KLN * BS * NHD, DH, DH, 1);
  k_perm_to_heads<<<g1(headsKL), 256, 0, stream>>>(tmp1, nullptr, Vh, KLN, 1);
  // ---- R head ----
  gNTbf(r_bf, HIDN, 0, WrT, HIDN, 0, tmp0, NDX, 0, nullptr, RL * BS, NDX, HIDN, 1);
  gNTbf(tmp0, DH, 0, w_rl, DH, 0, tmp1, DH, 0, rlin_b, RL * BS * NHD, DH, DH, 1);
  k_perm_to_heads<<<g1(headsR), 256, 0, stream>>>(tmp1, nullptr, Kr, RL, 1);
  // ---- Q head (h-stream) ----
  gNTbf(h_bf, HIDN, 0, WqT, HIDN, 0, tmp0, NDX, 0, nullptr, QL * BS, NDX, HIDN, 1);
  gNTbf(tmp0, DH, 0, w_ql, DH, 0, tmp1, DH, 0, qlin_b, QL * BS * NHD, DH, DH, 1);
  k_ef1<<<g1(BS * NHD * QL * 2), 256, 0, stream>>>(tmp1, rs_b, se, ef1h, QL, 1);
  k_perm_to_heads<<<g1(headsQ), 256, 0, stream>>>(tmp1, rw_b, Qh_rw, QL, 1);
  k_perm_to_heads<<<g1(headsQ), 256, 0, stream>>>(tmp1, rr_b, Qh_rr, QL, 1);
  // ---- Q head (g-stream): project, glin, one-hot gather ----
  gNTbf(g_bf, HIDN, 0, WqT, HIDN, 0, tmp0, NDX, 0, nullptr, QL * BS, NDX, HIDN, 1);
  gNTbf(tmp0, DH, 0, w_gl, DH, 0, tmp1, DH, 0, glin_b, QL * BS * NHD, DH, DH, 1);
  gNNbf(tmT_bf, QL, (long long)NPD * QL, tmp1, BS * NDX, NDX,
        qgath, NDX, (long long)NPD * NDX, nullptr, NPD, NDX, QL, BS);
  k_ef1<<<g1(BS * NHD * NPD * 2), 256, 0, stream>>>(qgath, rs_b, se, ef1g, NPD, 0);
  k_perm_to_heads<<<g1(headsNP), 256, 0, stream>>>(qgath, rw_b, Qg_rw, NPD, 0);
  k_perm_to_heads<<<g1(headsNP), 256, 0, stream>>>(qgath, rr_b, Qg_rr, NPD, 0);
  // ---- seg_mat gather for g: tm^T x seg_h ----
  gNNbf(tmT_bf, QL, (long long)NPD * QL, seg_h, 2 * KLN, (long long)QL * 2 * KLN,
        seg_g, 2 * KLN, (long long)NPD * 2 * KLN, nullptr, NPD, 2 * KLN, QL, BS);

  // =========================== h-stream ===========================
  gNTf(Qh_rw, DH, (long long)QL * DH, Kh, DH, (long long)KLN * DH,
       ac, KLN, (long long)QL * KLN, nullptr, QL, KLN, DH, BS * NHD);
  gNTf(Qh_rr, DH, (long long)QL * DH, Kr, DH, (long long)RL * DH,
       bd, RL, (long long)QL * RL, nullptr, QL, RL, DH, BS * NHD);
  k_softmax<<<dim3(QL, NHD, BS), 256, 0, stream>>>(ac, bd, ef1h, seg_h, mask_h, prob, QL);
  gNNbf(prob, KLN, (long long)QL * KLN, Vh, DH, (long long)KLN * DH,
        tmp0, DH, (long long)QL * DH, nullptr, QL, DH, KLN, BS * NHD);
  gNTbf(tmp0, DH, 0, w_dn, DH, 0, tmp1, DH, 0, dense_b, BS * NHD * QL, DH, DH, 1);
  k_perm_from_heads<<<g1(headsQ), 256, 0, stream>>>(tmp1, prob, QL);   // av3 -> prob buffer
  gNTf(prob, NDX, (long long)QL * NDX, w_po, NDX, 0,
       ao, HIDN, (long long)QL * HIDN, nullptr, QL, HIDN, NDX, BS);
  k_ln<<<dim3(QL * BS), 256, 0, stream>>>(ao, h_in, ln_gam, ln_bet, out, QL);

  // =========================== g-stream ===========================
  gNTf(Qg_rw, DH, (long long)NPD * DH, Kh, DH, (long long)KLN * DH,
       ac, KLN, (long long)NPD * KLN, nullptr, NPD, KLN, DH, BS * NHD);
  gNTf(Qg_rr, DH, (long long)NPD * DH, Kr, DH, (long long)RL * DH,
       bd, RL, (long long)NPD * RL, nullptr, NPD, RL, DH, BS * NHD);
  k_softmax<<<dim3(NPD, NHD, BS), 256, 0, stream>>>(ac, bd, ef1g, seg_g, mask_g, prob, NPD);
  gNNbf(prob, KLN, (long long)NPD * KLN, Vh, DH, (long long)KLN * DH,
        tmp0, DH, (long long)NPD * DH, nullptr, NPD, DH, KLN, BS * NHD);
  gNTbf(tmp0, DH, 0, w_dn, DH, 0, tmp1, DH, 0, dense_b, BS * NHD * NPD, DH, DH, 1);
  k_perm_from_heads<<<g1(headsNP), 256, 0, stream>>>(tmp1, qgath, NPD);  // av3_g -> qgath
  // one-hot scatter back to full sequence: tm x av3_g -> av_full (prob buffer)
  gNNbf(tm_bf, NPD, (long long)QL * NPD, qgath, NDX, (long long)NPD * NDX,
        prob, NDX, (long long)QL * NDX, nullptr, QL, NDX, NPD, BS);
  gNTf(prob, NDX, (long long)QL * NDX, w_po, NDX, 0,
       ao, HIDN, (long long)QL * HIDN, nullptr, QL, HIDN, NDX, BS);
  k_ln<<<dim3(QL * BS), 256, 0, stream>>>(ao, g_in, ln_gam, ln_bet,
                                          out + (long long)QL * BS * HIDN, QL);
}
